// REINFORCE_34308198760574
// MI455X (gfx1250) — matrix operand order repaired and checked by reading, not compiled, not run on hardware
//
#include <hip/hip_runtime.h>

#define HDIM 256
#define NCAND 100000
#define ROWS 64
#define LEAKY 0.01f

typedef unsigned int u32;
typedef unsigned short u16;
typedef __bf16 bf16_t;
typedef u32 u32x4 __attribute__((ext_vector_type(4)));
typedef float f32x4 __attribute__((ext_vector_type(4)));
typedef float f32x2 __attribute__((ext_vector_type(2)));
typedef bf16_t v16bf __attribute__((ext_vector_type(16)));
typedef bf16_t v2bf __attribute__((ext_vector_type(2)));
typedef float v8f __attribute__((ext_vector_type(8)));
typedef int i32x4v __attribute__((vector_size(16)));   // matches builtin param type

#if __has_builtin(__builtin_amdgcn_global_load_async_to_lds_b128) && \
    __has_builtin(__builtin_amdgcn_s_wait_asynccnt)
#define USE_ASYNC_GATHER 1
#else
#define USE_ASYNC_GATHER 0
#endif

union FragU { u32x4 q[2]; u32 u[8]; v16bf v; };

static __device__ __forceinline__ u16 f32_to_bf16(float f) {
  union { float f; u32 u; } v; v.f = f;
  u32 r = v.u + 0x7FFFu + ((v.u >> 16) & 1u);
  return (u16)(r >> 16);
}

#if __has_builtin(__builtin_amdgcn_cvt_pk_bf16_f32)
static __device__ __forceinline__ u32 pack_bf16x2(float lo, float hi) {
  union { v2bf b; u32 u; } c;
  c.b = __builtin_amdgcn_cvt_pk_bf16_f32(lo, hi);
  return c.u;
}
#else
static __device__ __forceinline__ u32 pack_bf16x2(float lo, float hi) {
  return (u32)f32_to_bf16(lo) | ((u32)f32_to_bf16(hi) << 16);
}
#endif

static __device__ __forceinline__ u16 bf16_of(float v) {
  return (u16)(pack_bf16x2(v, 0.f) & 0xFFFFu);
}

// ---------------------------------------------------------------------------
// Pack W (f32, row-major [K][256]) into WMMA B-fragment order, bf16.
// Fragment (kb, nt): 32 lanes x 16 bf16. lane -> N = nt*16 + (lane&15);
// K within 32-block: (e>>3)*16 + half*8 + (e&7)  (half = lane>>4, e = element 0..15).
// ---------------------------------------------------------------------------
__global__ __launch_bounds__(256) void k_pack_weights(
    const float* __restrict__ W1, const float* __restrict__ W2,
    const float* __restrict__ W3, u16* __restrict__ W1f,
    u16* __restrict__ W2f, u16* __restrict__ W3f) {
  int id = blockIdx.x * 256 + threadIdx.x;   // 0 .. 262143
  const float* src; u16* dst; int lid;
  if (id < 131072)            { src = W1; dst = W1f; lid = id; }
  else if (id < 131072+65536) { src = W2; dst = W2f; lid = id - 131072; }
  else                        { src = W3; dst = W3f; lid = id - 196608; }
  int e    = lid & 15;
  int lane = (lid >> 4) & 31;
  int frag = lid >> 9;
  int nt   = frag & 15;
  int kb   = frag >> 4;
  int k = kb * 32 + (e >> 3) * 16 + (lane >> 4) * 8 + (e & 7);
  int n = nt * 16 + (lane & 15);
  dst[lid] = f32_to_bf16(src[k * 256 + n]);
}

// A fragment (16x32 bf16) from row-major bf16 LDS buffer (ISA 16-bit A table).
static __device__ __forceinline__ v16bf load_a_frag(const u32* lds, int rowBaseU,
                                                    int kb, int half4) {
  FragU f;
  int base = rowBaseU + kb * 16 + half4;
#pragma unroll
  for (int r = 0; r < 8; ++r)
    f.u[r] = lds[base + ((r >> 2) << 3) + (r & 3)];
  return f.v;
}

// A fragment from row-major f32 LDS buffer, converting to bf16 on the fly.
static __device__ __forceinline__ v16bf load_a_frag_f32(const float* lds,
                                                        int rowBaseF, int kb,
                                                        int half) {
  FragU f;
  int base = rowBaseF + kb * 32 + half * 8;
#pragma unroll
  for (int r = 0; r < 8; ++r) {
    int kl = ((r >> 2) << 4) + ((r & 3) << 1);
    f32x2 p = *(const f32x2*)(lds + base + kl);
    f.u[r] = pack_bf16x2(p.x, p.y);
  }
  return f.v;
}

static __device__ __forceinline__ v16bf load_b_frag(const u16* __restrict__ wf,
                                                    int kb, int ntg, int lane) {
  const u32x4* p = (const u32x4*)(wf + ((size_t)((kb * 16 + ntg) * 32 + lane) << 4));
  FragU f;
  f.q[0] = p[0];
  f.q[1] = p[1];
  return f.v;
}

static __device__ __forceinline__ void gemm_tiles(const u32* aLds, int aStrideU,
                                                  int aRow, int nkb,
                                                  const u16* __restrict__ wf,
                                                  int colh, int lane, int half4,
                                                  v8f* acc) {
  int rowBaseU = aRow * aStrideU;
  for (int kb = 0; kb < nkb; ++kb) {
    v16bf a = load_a_frag(aLds, rowBaseU, kb, half4);
#pragma unroll
    for (int nt = 0; nt < 8; ++nt) {
      v16bf b = load_b_frag(wf, kb, colh * 8 + nt, lane);
      acc[nt] = __builtin_amdgcn_wmma_f32_16x16x32_bf16(
          false, a, false, b, (short)0, acc[nt], false, false);
    }
    if (kb + 1 < nkb)
      __builtin_prefetch(wf + ((size_t)(((kb + 1) * 16 + colh * 8) * 32 + lane) << 4), 0, 0);
  }
}

// ---------------------------------------------------------------------------
// Main kernel: 64 rows per block, 256 threads (8 waves).
// wave = 4 row-groups (16 rows) x 2 column halves (128 cols).
// LDS: feat f32 [64][512] @0 (128KB)  | h2 bf16 aliases feat after layer 1
//      h1   bf16 [64][256] @131072 (32KB)
//      sp   f32  [64][2]   @163840, sc f32 [64] @164352  -> 164608 bytes
// ---------------------------------------------------------------------------
__global__ __launch_bounds__(256) void k_mlp_scores(
    const float* __restrict__ m_emb, const float* __restrict__ op_emb,
    const float* __restrict__ job_srpt, const float* __restrict__ proc_norm,
    const int* __restrict__ m_ids, const int* __restrict__ op_ids,
    const int* __restrict__ job_ids,
    const float* __restrict__ W1, const float* __restrict__ b1,
    const float* __restrict__ b2, const float* __restrict__ b3,
    const float* __restrict__ Wout, const float* __restrict__ bout,
    const u16* __restrict__ W1f, const u16* __restrict__ W2f,
    const u16* __restrict__ W3f, float* __restrict__ scores) {
  extern __shared__ __align__(16) char smem[];
  float* featF = (float*)smem;              // 64*512 f32
  u16* h1      = (u16*)(smem + 131072);
  u16* h2      = (u16*)smem;                // aliases feat (dead after L1)
  float* sp    = (float*)(smem + 163840);
  float* sc    = (float*)(smem + 164352);

  int tid = threadIdx.x;
  int row0 = blockIdx.x * ROWS;

  // ---- gather features into LDS (f32) via async DMA ----
  for (int i = tid; i < ROWS * 128; i += 256) {
    int rl = i >> 7;
    int seg = i & 127;
    int g = row0 + rl; if (g >= NCAND) g = NCAND - 1;
    int base = seg << 2;
    const float* srcp;
    if (base < 256) srcp = m_emb + (size_t)m_ids[g] * 256 + base;
    else            srcp = op_emb + (size_t)op_ids[g] * 256 + (base - 256);
#if USE_ASYNC_GATHER
    __builtin_amdgcn_global_load_async_to_lds_b128(
        (__attribute__((address_space(1))) i32x4v*)srcp,
        (__attribute__((address_space(3))) i32x4v*)(featF + rl * 512 + base),
        0, 0);
#else
    f32x4 v = *(const f32x4*)srcp;
    *(f32x4*)(featF + rl * 512 + base) = v;
#endif
  }
  if (tid < ROWS) {
    int g = row0 + tid; if (g >= NCAND) g = NCAND - 1;
    sp[tid * 2]     = job_srpt[job_ids[g]];
    sp[tid * 2 + 1] = proc_norm[g];
    sc[tid] = bout[0];
  }
#if USE_ASYNC_GATHER
  __builtin_amdgcn_s_wait_asynccnt(0);
#endif
  __syncthreads();

  int lane = tid & 31;
  int wave = tid >> 5;
  int rowg = wave >> 1;
  int colh = wave & 1;
  int half = lane >> 4;
  int half4 = half << 2;
  int aRow = rowg * 16 + (lane & 15);

  v8f acc[8];
#pragma unroll
  for (int t = 0; t < 8; ++t)
#pragma unroll
    for (int j = 0; j < 8; ++j) acc[t][j] = 0.0f;

  // ---- layer 1: feat[64x512] x W1[512x256] (+ rank-1 scalar features) ----
  {
    int rowBaseF = aRow * 512;
    for (int kb = 0; kb < 16; ++kb) {
      v16bf a = load_a_frag_f32(featF, rowBaseF, kb, half);
#pragma unroll
      for (int nt = 0; nt < 8; ++nt) {
        v16bf b = load_b_frag(W1f, kb, colh * 8 + nt, lane);
        acc[nt] = __builtin_amdgcn_wmma_f32_16x16x32_bf16(
            false, a, false, b, (short)0, acc[nt], false, false);
      }
      if (kb + 1 < 16)
        __builtin_prefetch(W1f + ((size_t)(((kb + 1) * 16 + colh * 8) * 32 + lane) << 4), 0, 0);
    }
  }

  float sr[8], pr[8];
#pragma unroll
  for (int r = 0; r < 8; ++r) {
    int ml = rowg * 16 + r + (half << 3);
    sr[r] = sp[ml * 2];
    pr[r] = sp[ml * 2 + 1];
  }
#pragma unroll
  for (int nt = 0; nt < 8; ++nt) {
    int n = colh * 128 + nt * 16 + (lane & 15);
    float bb  = b1[n];
    float w5a = W1[512 * 256 + n];
    float w5b = W1[513 * 256 + n];
#pragma unroll
    for (int r = 0; r < 8; ++r) {
      int ml = rowg * 16 + r + (half << 3);
      float v = acc[nt][r] + bb + sr[r] * w5a + pr[r] * w5b;
      v = v > 0.f ? v : LEAKY * v;
      h1[ml * 256 + n] = bf16_of(v);
    }
  }
  __syncthreads();

  // ---- layer 2: h1 x W2 ----
#pragma unroll
  for (int t = 0; t < 8; ++t)
#pragma unroll
    for (int j = 0; j < 8; ++j) acc[t][j] = 0.0f;
  gemm_tiles((const u32*)h1, 128, aRow, 8, W2f, colh, lane, half4, acc);
#pragma unroll
  for (int nt = 0; nt < 8; ++nt) {
    int n = colh * 128 + nt * 16 + (lane & 15);
    float bb = b2[n];
#pragma unroll
    for (int r = 0; r < 8; ++r) {
      int ml = rowg * 16 + r + (half << 3);
      float v = acc[nt][r] + bb;
      v = v > 0.f ? v : LEAKY * v;
      h2[ml * 256 + n] = bf16_of(v);
    }
  }
  __syncthreads();

  // ---- layer 3: h2 x W3, fused output dot with Wout ----
#pragma unroll
  for (int t = 0; t < 8; ++t)
#pragma unroll
    for (int j = 0; j < 8; ++j) acc[t][j] = 0.0f;
  gemm_tiles((const u32*)h2, 128, aRow, 8, W3f, colh, lane, half4, acc);

  float part[8];
#pragma unroll
  for (int r = 0; r < 8; ++r) part[r] = 0.0f;
#pragma unroll
  for (int nt = 0; nt < 8; ++nt) {
    int n = colh * 128 + nt * 16 + (lane & 15);
    float bb = b3[n];
    float wv = Wout[n];
#pragma unroll
    for (int r = 0; r < 8; ++r) {
      float v = acc[nt][r] + bb;
      v = v > 0.f ? v : LEAKY * v;
      part[r] += v * wv;
    }
  }
#pragma unroll
  for (int r = 0; r < 8; ++r) {
    int ml = rowg * 16 + r + (half << 3);
    atomicAdd(&sc[ml], part[r]);
  }
  __syncthreads();
  if (tid < ROWS) {
    int g = row0 + tid;
    if (g < NCAND) scores[g] = sc[tid];
  }
}

// ---------------------------------------------------------------------------
// Reductions: argmax, log-sum-exp, entropy, probs
// ---------------------------------------------------------------------------
__global__ __launch_bounds__(256) void k_argmax_part(const float* __restrict__ s,
                                                     int n, float* pmax, int* pidx) {
  __shared__ float sm[256];
  __shared__ int si[256];
  int tid = threadIdx.x;
  float best = -3.4e38f; int bi = 0x7FFFFFFF;
  for (int i = blockIdx.x * 256 + tid; i < n; i += gridDim.x * 256) {
    float v = s[i];
    if (v > best || (v == best && i < bi)) { best = v; bi = i; }
  }
  sm[tid] = best; si[tid] = bi;
  __syncthreads();
  for (int o = 128; o > 0; o >>= 1) {
    if (tid < o) {
      float v = sm[tid + o]; int j = si[tid + o];
      if (v > sm[tid] || (v == sm[tid] && j < si[tid])) { sm[tid] = v; si[tid] = j; }
    }
    __syncthreads();
  }
  if (tid == 0) { pmax[blockIdx.x] = sm[0]; pidx[blockIdx.x] = si[0]; }
}

__global__ __launch_bounds__(256) void k_argmax_fin(const float* __restrict__ pmax,
                                                    const int* __restrict__ pidx,
                                                    float* gmax, int* gidx,
                                                    float* out_idx) {
  __shared__ float sm[256];
  __shared__ int si[256];
  int tid = threadIdx.x;
  sm[tid] = pmax[tid]; si[tid] = pidx[tid];
  __syncthreads();
  for (int o = 128; o > 0; o >>= 1) {
    if (tid < o) {
      float v = sm[tid + o]; int j = si[tid + o];
      if (v > sm[tid] || (v == sm[tid] && j < si[tid])) { sm[tid] = v; si[tid] = j; }
    }
    __syncthreads();
  }
  if (tid == 0) { *gmax = sm[0]; *gidx = si[0]; *out_idx = (float)si[0]; }
}

__global__ __launch_bounds__(256) void k_sum_part(const float* __restrict__ s, int n,
                                                  const float* __restrict__ gmax,
                                                  float* p0, float* p1) {
  __shared__ float s0m[256];
  __shared__ float s1m[256];
  int tid = threadIdx.x;
  float mx = *gmax;
  float s0 = 0.f, s1 = 0.f;
  for (int i = blockIdx.x * 256 + tid; i < n; i += gridDim.x * 256) {
    float t = s[i] - mx;
    float e = __expf(t);
    s0 += e; s1 += t * e;
  }
  s0m[tid] = s0; s1m[tid] = s1;
  __syncthreads();
  for (int o = 128; o > 0; o >>= 1) {
    if (tid < o) { s0m[tid] += s0m[tid + o]; s1m[tid] += s1m[tid + o]; }
    __syncthreads();
  }
  if (tid == 0) { p0[blockIdx.x] = s0m[0]; p1[blockIdx.x] = s1m[0]; }
}

__global__ __launch_bounds__(256) void k_sum_fin(const float* __restrict__ p0,
                                                 const float* __restrict__ p1,
                                                 float* zOut, float* outLogp,
                                                 float* outEnt) {
  __shared__ float s0m[256];
  __shared__ float s1m[256];
  int tid = threadIdx.x;
  s0m[tid] = p0[tid]; s1m[tid] = p1[tid];
  __syncthreads();
  for (int o = 128; o > 0; o >>= 1) {
    if (tid < o) { s0m[tid] += s0m[tid + o]; s1m[tid] += s1m[tid + o]; }
    __syncthreads();
  }
  if (tid == 0) {
    float Z = s0m[0];
    float lz = logf(Z);
    *zOut = Z;
    *outLogp = -lz;                 // log prob of argmax: t=0 there
    *outEnt  = lz - s1m[0] / Z;     // -sum p log p
  }
}

__global__ __launch_bounds__(256) void k_probs(const float* __restrict__ s, int n,
                                               const float* __restrict__ gmax,
                                               const float* __restrict__ z,
                                               float* __restrict__ probs) {
  int i = blockIdx.x * 256 + threadIdx.x;
  if (i < n) probs[i] = __expf(s[i] - *gmax) / *z;
}

// ---------------------------------------------------------------------------
extern "C" void kernel_launch(void* const* d_in, const int* in_sizes, int n_in,
                              void* d_out, int out_size, void* d_ws, size_t ws_size,
                              hipStream_t stream) {
  (void)in_sizes; (void)n_in; (void)out_size; (void)ws_size;
  const float* m_emb     = (const float*)d_in[0];
  const float* op_emb    = (const float*)d_in[1];
  const float* job_srpt  = (const float*)d_in[2];
  const float* proc_norm = (const float*)d_in[3];
  const int*   m_ids     = (const int*)d_in[4];
  const int*   op_ids    = (const int*)d_in[5];
  const int*   job_ids   = (const int*)d_in[6];
  const float* W1   = (const float*)d_in[7];
  const float* b1   = (const float*)d_in[8];
  const float* W2   = (const float*)d_in[9];
  const float* b2   = (const float*)d_in[10];
  const float* W3   = (const float*)d_in[11];
  const float* b3   = (const float*)d_in[12];
  const float* Wout = (const float*)d_in[13];
  const float* bout = (const float*)d_in[14];

  char* ws = (char*)d_ws;
  u16* W1f      = (u16*)(ws);
  u16* W2f      = (u16*)(ws + 262144);
  u16* W3f      = (u16*)(ws + 393216);
  float* scores = (float*)(ws + 524288);
  float* pmax   = (float*)(ws + 924288);
  int*   pidx   = (int*)(ws + 925312);
  float* gmax   = (float*)(ws + 926336);
  int*   gidx   = (int*)(ws + 926340);
  float* p0     = (float*)(ws + 926344);
  float* p1     = (float*)(ws + 927368);
  float* zv     = (float*)(ws + 928392);
  float* out    = (float*)d_out;

  k_pack_weights<<<1024, 256, 0, stream>>>(W1, W2, W3, W1f, W2f, W3f);

  int nblk = (NCAND + ROWS - 1) / ROWS;
  size_t shbytes = 164608;
  k_mlp_scores<<<nblk, 256, shbytes, stream>>>(
      m_emb, op_emb, job_srpt, proc_norm, m_ids, op_ids, job_ids,
      W1, b1, b2, b3, Wout, bout, W1f, W2f, W3f, scores);

  k_argmax_part<<<256, 256, 0, stream>>>(scores, NCAND, pmax, pidx);
  k_argmax_fin<<<1, 256, 0, stream>>>(pmax, pidx, gmax, gidx, out + NCAND);
  k_sum_part<<<256, 256, 0, stream>>>(scores, NCAND, gmax, p0, p1);
  k_sum_fin<<<1, 256, 0, stream>>>(p0, p1, zv, out + NCAND + 1, out + NCAND + 2);
  k_probs<<<(NCAND + 255) / 256, 256, 0, stream>>>(scores, NCAND, gmax, zv, out);
}
